// AdaptiveEmbeddingT2I_87746181857677
// MI455X (gfx1250) — matrix operand order repaired and checked by reading, not compiled, not run on hardware
//
#include <hip/hip_runtime.h>
#include <hip/hip_bf16.h>
#include <math.h>

typedef __attribute__((ext_vector_type(16))) _Float16 v16h;
typedef __attribute__((ext_vector_type(8)))  float    v8f;

#define Bi 64
#define R  36
#define D  1024
#define Bc 32
#define T  64
#define H  128
#define GAMMA_SM 10.0f

// ---------------------------------------------------------------------------
// K1: ragged mean pool over captions + fused L2 norm.
//   cap_embed (Bc,T,D) -> cap_repr (Bc,D), cap_vec = l2norm(cap_repr)
// one block per caption, 256 threads, each owns 4 channels.
// ---------------------------------------------------------------------------
__global__ void cap_repr_kernel(const float* __restrict__ cap,
                                const int* __restrict__ lens,
                                float* __restrict__ capr,
                                float* __restrict__ capv) {
    const int c   = blockIdx.x;
    const int tid = threadIdx.x;
    __shared__ float red[256];

    const int   L    = lens[c];
    const float invL = 1.0f / (float)L;

    float vals[4];
    float ss = 0.0f;
#pragma unroll
    for (int k = 0; k < 4; ++k) {
        const int d = tid + k * 256;
        float s = 0.0f;
        for (int j = 0; j < T; ++j) {
            float x = cap[(c * T + j) * D + d];
            s += (j < L) ? x : 0.0f;
        }
        s *= invL;
        vals[k] = s;
        capr[c * D + d] = s;
        ss += s * s;
    }
    red[tid] = ss;
    __syncthreads();
    for (int off = 128; off > 0; off >>= 1) {
        if (tid < off) red[tid] += red[tid + off];
        __syncthreads();
    }
    const float inv = 1.0f / (sqrtf(red[0]) + 1e-8f);
#pragma unroll
    for (int k = 0; k < 4; ++k) {
        const int d = tid + k * 256;
        capv[c * D + d] = vals[k] * inv;
    }
}

// ---------------------------------------------------------------------------
// K2: BatchNorm batch stats per channel d over (Bi*R)=2304 samples.
// img (Bi,R,D) contiguous -> sample i lives at i*D + d (coalesced over d).
// ---------------------------------------------------------------------------
__global__ void bn_stats_kernel(const float* __restrict__ img,
                                float* __restrict__ mean,
                                float* __restrict__ invstd) {
    const int d = blockIdx.x * 256 + threadIdx.x;
    float s = 0.0f, ss = 0.0f;
    for (int i = 0; i < Bi * R; ++i) {
        float x = img[i * D + d];
        s += x;
        ss += x * x;
    }
    const float m = s * (1.0f / (Bi * R));
    const float v = ss * (1.0f / (Bi * R)) - m * m;
    mean[d]   = m;
    invstd[d] = rsqrtf(v + 1e-5f);
}

// ---------------------------------------------------------------------------
// K3: conditioning MLP with WMMA f16 (f32 accumulate).
//   hidden = relu(cap_repr @ W1 + b1)   M=32 K=1024 N=128
//   out    = hidden   @ W2 + b2         M=32 K=128  N=1024
// grid = 2 blocks (gamma / beta), 256 threads = 8 waves.
// Fragment layouts per CDNA5 ISA 7.12.2 (wave32):
//   A 16x32 f16: lane = M%16 (+half selects K+8/K+24 block); pairs per dword
//   B 32x16 f16: N = lane&15, K = (elem&7) + (lane>=16)*8 + (elem>>3)*16
//   C/D f32:     M = elem + 8*(lane>=16), N = lane&15
// ---------------------------------------------------------------------------
__global__ void mlp_wmma_kernel(const float* __restrict__ capr,
                                const float* __restrict__ Wg1, const float* __restrict__ bg1,
                                const float* __restrict__ Wg2, const float* __restrict__ bg2,
                                const float* __restrict__ Wb1, const float* __restrict__ bb1,
                                const float* __restrict__ Wb2, const float* __restrict__ bb2,
                                float* __restrict__ gammas, float* __restrict__ betas) {
    const bool isBeta = (blockIdx.x != 0);
    const float* W1 = isBeta ? Wb1 : Wg1;
    const float* B1 = isBeta ? bb1 : bg1;
    const float* W2 = isBeta ? Wb2 : Wg2;
    const float* B2 = isBeta ? bb2 : bg2;
    float* out = isBeta ? betas : gammas;

    __shared__ _Float16 hidden[32][H];  // 8 KB

    const int tid   = threadIdx.x;
    const int wave  = tid >> 5;
    const int lane  = tid & 31;
    const int lhalf = lane >> 4;   // 0 or 1
    const int l16   = lane & 15;

    // ---------------- Stage 1: hidden (2 m-tiles x 8 n-tiles, K=1024) ------
    {
        const int t0    = wave * 2;        // each wave: two n-tiles, same m-tile
        const int mtile = t0 >> 3;
        const int n0    = t0 & 7;
        const int n1    = n0 + 1;
        v8f acc0 = {};
        v8f acc1 = {};
        const int arow = (mtile * 16 + l16) * D;
        for (int kt = 0; kt < 32; ++kt) {       // K/32 steps
            const int kbase = kt * 32;
            v16h a;
#pragma unroll
            for (int v = 0; v < 8; ++v) {
                const int K = kbase + ((v & 3) * 2) + ((v >> 2) * 16) + lhalf * 8;
                const float* p = capr + arow + K;
                a[2 * v]     = (_Float16)p[0];
                a[2 * v + 1] = (_Float16)p[1];
            }
            v16h b0v, b1v;
#pragma unroll
            for (int e = 0; e < 16; ++e) {
                const int K = kbase + (e & 7) + lhalf * 8 + (e >> 3) * 16;
                b0v[e] = (_Float16)W1[K * H + n0 * 16 + l16];
                b1v[e] = (_Float16)W1[K * H + n1 * 16 + l16];
            }
            acc0 = __builtin_amdgcn_wmma_f32_16x16x32_f16(false, a, false, b0v,
                                                          (short)0, acc0, false, false);
            acc1 = __builtin_amdgcn_wmma_f32_16x16x32_f16(false, a, false, b1v,
                                                          (short)0, acc1, false, false);
        }
        // epilogue: +bias, relu, f16 -> LDS
#pragma unroll
        for (int r = 0; r < 8; ++r) {
            const int row = mtile * 16 + r + lhalf * 8;
            const int c0  = n0 * 16 + l16;
            const int c1  = n1 * 16 + l16;
            float h0 = acc0[r] + B1[c0]; h0 = h0 > 0.0f ? h0 : 0.0f;
            float h1 = acc1[r] + B1[c1]; h1 = h1 > 0.0f ? h1 : 0.0f;
            hidden[row][c0] = (_Float16)h0;
            hidden[row][c1] = (_Float16)h1;
        }
    }
    __syncthreads();

    // ---------------- Stage 2: out (2 m-tiles x 64 n-tiles, K=128) ---------
    {
        const int mtile = wave >> 2;          // waves 0-3: m=0, 4-7: m=1
        const int nbase = (wave & 3) * 16;    // 16 n-tiles per wave
        // preload all 4 K-step A fragments from LDS (reused for 16 n-tiles)
        v16h afr[4];
        const int arow = mtile * 16 + l16;
#pragma unroll
        for (int kt = 0; kt < 4; ++kt) {
#pragma unroll
            for (int v = 0; v < 8; ++v) {
                const int K = kt * 32 + ((v & 3) * 2) + ((v >> 2) * 16) + lhalf * 8;
                afr[kt][2 * v]     = hidden[arow][K];
                afr[kt][2 * v + 1] = hidden[arow][K + 1];
            }
        }
        for (int i = 0; i < 16; ++i) {
            const int nt = nbase + i;
            v8f acc = {};
#pragma unroll
            for (int kt = 0; kt < 4; ++kt) {
                v16h bfr;
#pragma unroll
                for (int e = 0; e < 16; ++e) {
                    const int K = kt * 32 + (e & 7) + lhalf * 8 + (e >> 3) * 16;
                    bfr[e] = (_Float16)W2[K * D + nt * 16 + l16];
                }
                acc = __builtin_amdgcn_wmma_f32_16x16x32_f16(false, afr[kt], false, bfr,
                                                             (short)0, acc, false, false);
            }
#pragma unroll
            for (int r = 0; r < 8; ++r) {
                const int row = mtile * 16 + r + lhalf * 8;
                const int col = nt * 16 + l16;
                out[row * D + col] = acc[r] + B2[col];
            }
        }
    }
}

// ---------------------------------------------------------------------------
// K4: fused BN-apply + film + softmax(x*10) over R + weighted mean + l2norm
//     + dot with cap_vec. One block per (c,b); img slice re-read from L2.
//   sims[b,c] = (sum_d v_d * capv[c,d]) / (||v|| + 1e-8)
//   v_d = (1/R) * sum_r softmax_r(10*o)_r * o_r,  o = bn*gamma + beta
// ---------------------------------------------------------------------------
__global__ void fused_kernel(const float* __restrict__ img,
                             const float* __restrict__ mean,
                             const float* __restrict__ invstd,
                             const float* __restrict__ gammas,
                             const float* __restrict__ betas,
                             const float* __restrict__ capv,
                             float* __restrict__ sims) {
    const int idx = blockIdx.x;   // 0..2047
    const int b   = idx & (Bi - 1);
    const int c   = idx >> 6;
    const int tid = threadIdx.x;

    __shared__ float redDot[256];
    __shared__ float redSS[256];

    const float* imgb = img + (size_t)b * R * D;
    float dot = 0.0f, ss = 0.0f;

#pragma unroll
    for (int k = 0; k < 4; ++k) {
        const int d  = tid + k * 256;
        const float g  = gammas[c * D + d];
        const float be = betas[c * D + d];
        const float mu = mean[d];
        const float is = invstd[d];
        // pass 1: row max of scaled values (stable softmax); L2-resident reads
        float mx = -1e30f;
        for (int r = 0; r < R; ++r) {
            float o  = (imgb[r * D + d] - mu) * is * g + be;
            float so = o * GAMMA_SM;
            mx = so > mx ? so : mx;
        }
        // pass 2: exp-sum and weighted numerator
        float se = 0.0f, num = 0.0f;
        for (int r = 0; r < R; ++r) {
            float o = (imgb[r * D + d] - mu) * is * g + be;
            float e = __expf(o * GAMMA_SM - mx);
            se  += e;
            num += e * o;
        }
        const float v = (num / se) * (1.0f / (float)R);
        ss  += v * v;
        dot += v * capv[c * D + d];
    }
    redDot[tid] = dot;
    redSS[tid]  = ss;
    __syncthreads();
    for (int off = 128; off > 0; off >>= 1) {
        if (tid < off) {
            redDot[tid] += redDot[tid + off];
            redSS[tid]  += redSS[tid + off];
        }
        __syncthreads();
    }
    if (tid == 0) {
        sims[b * Bc + c] = redDot[0] / (sqrtf(redSS[0]) + 1e-8f);
    }
}

// ---------------------------------------------------------------------------
extern "C" void kernel_launch(void* const* d_in, const int* in_sizes, int n_in,
                              void* d_out, int out_size, void* d_ws, size_t ws_size,
                              hipStream_t stream) {
    const float* img = (const float*)d_in[0];   // (64,36,1024)
    const float* cap = (const float*)d_in[1];   // (32,64,1024)
    const float* Wg1 = (const float*)d_in[2];   // (1024,128)
    const float* bg1 = (const float*)d_in[3];   // (128)
    const float* Wg2 = (const float*)d_in[4];   // (128,1024)
    const float* bg2 = (const float*)d_in[5];   // (1024)
    const float* Wb1 = (const float*)d_in[6];
    const float* bb1 = (const float*)d_in[7];
    const float* Wb2 = (const float*)d_in[8];
    const float* bb2 = (const float*)d_in[9];
    const int*  lens = (const int*)d_in[10];    // (32)
    float* sims = (float*)d_out;                // (64,32)

    // workspace layout (floats)
    float* ws      = (float*)d_ws;
    float* capr    = ws;                    // 32*1024
    float* capvec  = capr   + Bc * D;       // 32*1024
    float* mean    = capvec + Bc * D;       // 1024
    float* invstd  = mean   + D;            // 1024
    float* gammas  = invstd + D;            // 32*1024
    float* betas   = gammas + Bc * D;       // 32*1024

    cap_repr_kernel<<<Bc, 256, 0, stream>>>(cap, lens, capr, capvec);
    bn_stats_kernel<<<D / 256, 256, 0, stream>>>(img, mean, invstd);
    mlp_wmma_kernel<<<2, 256, 0, stream>>>(capr, Wg1, bg1, Wg2, bg2,
                                           Wb1, bb1, Wb2, bb2, gammas, betas);
    fused_kernel<<<Bc * Bi, 256, 0, stream>>>(img, mean, invstd, gammas, betas,
                                              capvec, sims);
}
